// MultiHeadSelfAttention_89953795047640
// MI455X (gfx1250) — compile-verified, matrix operand order checked by reading, not run on hardware
//
#include <hip/hip_runtime.h>
#include <hip/hip_bf16.h>

// ---------------------------------------------------------------------------
// Multi-head self-attention, MI455X (gfx1250, wave32).
// bf16 WMMA (v_wmma_f32_16x16x32_bf16) for all matmuls, f32 softmax.
// ---------------------------------------------------------------------------

typedef __bf16 bf16_t;
typedef bf16_t bf16x16 __attribute__((ext_vector_type(16)));
typedef float  f32x8   __attribute__((ext_vector_type(8)));
typedef unsigned int u32x4 __attribute__((ext_vector_type(4)));

union Frag { u32x4 u[2]; bf16x16 v; };

#define WMMA_BF16(A, B, C) \
  __builtin_amdgcn_wmma_f32_16x16x32_bf16(false, (A), false, (B), (short)0, (C), false, false)

__device__ __forceinline__ unsigned short f2bf(float f) {
  unsigned u = __builtin_bit_cast(unsigned, f);
  u += 0x7FFFu + ((u >> 16) & 1u);   // round-to-nearest-even
  return (unsigned short)(u >> 16);
}

__device__ __forceinline__ u32x4 ld4(const unsigned short* p) {
  return *reinterpret_cast<const u32x4*>(p);
}
__device__ __forceinline__ void st4(unsigned short* p, u32x4 v) {
  *reinterpret_cast<u32x4*>(p) = v;
}

static constexpr int Dm  = 1024;   // model dim
static constexpr int Hn  = 16;     // heads
static constexpr int HDm = 64;     // head dim
static constexpr int Sm  = 2048;   // sequence
static constexpr int Bm  = 2;      // batch
static constexpr int Mtot = Bm * Sm;  // 4096 rows

// ---------------------------------------------------------------------------
// Stage 1a: f32 -> bf16 elementwise convert
// ---------------------------------------------------------------------------
__global__ void cvt_f32_bf16(const float* __restrict__ in,
                             unsigned short* __restrict__ out) {
  int idx = blockIdx.x * blockDim.x + threadIdx.x;
  out[idx] = f2bf(in[idx]);
}

// Stage 1b: f32 [K][N] -> bf16 transposed [N][K]  (D x D)
__global__ void cvt_transpose_bf16(const float* __restrict__ in,
                                   unsigned short* __restrict__ out) {
  int idx = blockIdx.x * blockDim.x + threadIdx.x;   // 0 .. D*D-1
  int k = idx >> 10;
  int n = idx & (Dm - 1);
  out[n * Dm + k] = f2bf(in[idx]);                   // coalesced read
}

// ---------------------------------------------------------------------------
// Stage 2/4: WMMA GEMM.  C[m,n] = sum_k A[m,k]*WT[n,k] + bias[n]
//   A : bf16 [4096][1024] row-major, WT : bf16 [1024][1024] (pre-transposed W)
//   mode 0: store bf16 Q  [B,H,S,HD]
//   mode 1: store bf16 K  [B,H,S,HD]
//   mode 2: store bf16 Vt [B,H,HD,S]
//   mode 3: store f32  out[4096][1024]
// Block: 256 threads (8 waves). Tile 128(M) x 128(N), BK = 32.
// Wave w owns rows [w*16, w*16+16) x all 128 cols -> 8 accumulator tiles.
// ---------------------------------------------------------------------------
__global__ __launch_bounds__(256) void gemm_bf16_wmma(
    const unsigned short* __restrict__ A,
    const unsigned short* __restrict__ WT,
    const float* __restrict__ bias,
    void* __restrict__ outp, int mode) {

  __shared__ __align__(16) unsigned short Asm[128 * 40];  // rows padded to 40
  __shared__ __align__(16) unsigned short Bsm[128 * 40];

  const int tid  = threadIdx.x;
  const int wave = tid >> 5;
  const int lane = tid & 31;
  const int l15  = lane & 15;
  const int hi8  = (lane >> 4) * 8;        // 0 or 8
  const int off  = hi8;                    // fragment k-offset

  const int ntile = blockIdx.x * 128;
  const int mtile = blockIdx.y * 128;
  const int mrow  = wave * 16;

  f32x8 acc[8];
#pragma unroll
  for (int j = 0; j < 8; ++j)
#pragma unroll
    for (int i = 0; i < 8; ++i) acc[j][i] = 0.0f;

  for (int kb = 0; kb < Dm; kb += 32) {
    __syncthreads();
#pragma unroll
    for (int p = 0; p < 2; ++p) {
      int id2 = tid + 256 * p;             // 0..511
      int row = id2 >> 2;                  // 0..127
      int c8  = (id2 & 3) * 8;             // 0,8,16,24
      st4(&Asm[row * 40 + c8], ld4(&A[(size_t)(mtile + row) * Dm + kb + c8]));
      st4(&Bsm[row * 40 + c8], ld4(&WT[(size_t)(ntile + row) * Dm + kb + c8]));
    }
    if (kb + 32 < Dm) {                    // prefetch next K-slab
      __builtin_prefetch(&A[(size_t)(mtile + (tid >> 2)) * Dm + kb + 32], 0, 0);
      __builtin_prefetch(&WT[(size_t)(ntile + (tid >> 2)) * Dm + kb + 32], 0, 0);
    }
    __syncthreads();

    Frag af;
    af.u[0] = ld4(&Asm[(mrow + l15) * 40 + off]);
    af.u[1] = ld4(&Asm[(mrow + l15) * 40 + off + 16]);
#pragma unroll
    for (int j = 0; j < 8; ++j) {
      Frag bf_;
      bf_.u[0] = ld4(&Bsm[(j * 16 + l15) * 40 + off]);
      bf_.u[1] = ld4(&Bsm[(j * 16 + l15) * 40 + off + 16]);
      acc[j] = WMMA_BF16(af.v, bf_.v, acc[j]);
    }
  }

  // Epilogue: bias + store per mode
#pragma unroll
  for (int j = 0; j < 8; ++j) {
#pragma unroll
    for (int i = 0; i < 8; ++i) {
      int m = mtile + mrow + i + hi8;      // global row  (b*2048+s)
      int n = ntile + j * 16 + l15;        // global col  (h*64+d)
      float v = acc[j][i] + bias[n];
      if (mode == 3) {
        ((float*)outp)[(size_t)m * Dm + n] = v;
      } else {
        int b = m >> 11, s = m & (Sm - 1);
        int h = n >> 6,  d = n & (HDm - 1);
        int bh = b * Hn + h;
        unsigned short bv = f2bf(v);
        if (mode == 2) {                   // Vt [B,H,HD,S]
          ((unsigned short*)outp)[((size_t)bh * HDm + d) * Sm + s] = bv;
        } else {                           // Q/K [B,H,S,HD]
          ((unsigned short*)outp)[((size_t)bh * Sm + s) * HDm + d] = bv;
        }
      }
    }
  }
}

// ---------------------------------------------------------------------------
// Stage 3: flash-attention.  Block = (b, h, 64 Q rows), 128 threads = 4 waves.
// Wave owns a 16-row Q strip; loops over 64-wide K/V tiles with online softmax.
// ---------------------------------------------------------------------------
__global__ __launch_bounds__(128) void attn_wmma(
    const unsigned short* __restrict__ Q,    // [B,H,S,HD] bf16
    const unsigned short* __restrict__ K,    // [B,H,S,HD] bf16
    const unsigned short* __restrict__ Vt,   // [B,H,HD,S] bf16
    unsigned short* __restrict__ attnOut) {  // [B,S,H*HD] bf16

  __shared__ __align__(16) unsigned short Ksm[64 * 80];       // [t][d], pad 80
  __shared__ __align__(16) unsigned short Vsm[64 * 80];       // [d][t], pad 80
  __shared__ __align__(16) unsigned short Psm[4 * 16 * 80];   // per-wave P

  const int tid  = threadIdx.x;
  const int wave = tid >> 5;
  const int lane = tid & 31;
  const int l15  = lane & 15;
  const int hi8  = (lane >> 4) * 8;
  const int off  = hi8;

  const int bh = blockIdx.y;               // b*16 + h
  const int b  = bh >> 4;
  const int h  = bh & 15;
  const int sbase = blockIdx.x * 64 + wave * 16;

  const unsigned short* Qh = Q  + (size_t)bh * Sm * HDm;
  const unsigned short* Kh = K  + (size_t)bh * Sm * HDm;
  const unsigned short* Vh = Vt + (size_t)bh * HDm * Sm;
  unsigned short* Pw = &Psm[wave * 16 * 80];

  // Q strip fragments (16 x 64 = 2 k-steps), cached in registers
  bf16x16 qfrag[2];
#pragma unroll
  for (int ks = 0; ks < 2; ++ks) {
    Frag qa;
    const unsigned short* base = &Qh[(size_t)(sbase + l15) * HDm + ks * 32 + off];
    qa.u[0] = ld4(base);
    qa.u[1] = ld4(base + 16);
    qfrag[ks] = qa.v;
  }

  f32x8 Oacc[4];
  float m_i[8], l_i[8];
#pragma unroll
  for (int j = 0; j < 4; ++j)
#pragma unroll
    for (int i = 0; i < 8; ++i) Oacc[j][i] = 0.0f;
#pragma unroll
  for (int i = 0; i < 8; ++i) { m_i[i] = -3.0e38f; l_i[i] = 0.0f; }

  for (int tb = 0; tb < Sm; tb += 64) {
    __syncthreads();
#pragma unroll
    for (int p = 0; p < 4; ++p) {
      int id2 = tid + 128 * p;             // 0..511
      int row = id2 >> 3;                  // 0..63
      int c8  = (id2 & 7) * 8;             // 0..56
      st4(&Ksm[row * 80 + c8], ld4(&Kh[(size_t)(tb + row) * HDm + c8]));
      st4(&Vsm[row * 80 + c8], ld4(&Vh[(size_t)row * Sm + tb + c8]));
    }
    __syncthreads();

    // S = Q * K^T  (16 x 64 strip, 4 col-subtiles x 2 k-steps)
    f32x8 Sacc[4];
#pragma unroll
    for (int j = 0; j < 4; ++j) {
#pragma unroll
      for (int i = 0; i < 8; ++i) Sacc[j][i] = 0.0f;
#pragma unroll
      for (int ks = 0; ks < 2; ++ks) {
        Frag kb_;
        int trow = j * 16 + l15;
        int dof  = ks * 32 + off;
        kb_.u[0] = ld4(&Ksm[trow * 80 + dof]);
        kb_.u[1] = ld4(&Ksm[trow * 80 + dof + 16]);
        Sacc[j] = WMMA_BF16(qfrag[ks], kb_.v, Sacc[j]);
      }
    }

    // Online softmax per row (8 rows per lane; 16-lane half-wave reductions)
#pragma unroll
    for (int i = 0; i < 8; ++i) {
      float sv[4];
      float mx = -3.0e38f;
#pragma unroll
      for (int j = 0; j < 4; ++j) { sv[j] = Sacc[j][i] * 0.125f; mx = fmaxf(mx, sv[j]); }
#pragma unroll
      for (int msk = 1; msk < 16; msk <<= 1) mx = fmaxf(mx, __shfl_xor(mx, msk, 32));
      float mnew  = fmaxf(m_i[i], mx);
      float alpha = __expf(m_i[i] - mnew);
      float pj[4], psum = 0.0f;
#pragma unroll
      for (int j = 0; j < 4; ++j) { pj[j] = __expf(sv[j] - mnew); psum += pj[j]; }
#pragma unroll
      for (int msk = 1; msk < 16; msk <<= 1) psum += __shfl_xor(psum, msk, 32);
      l_i[i] = l_i[i] * alpha + psum;
      m_i[i] = mnew;
#pragma unroll
      for (int j = 0; j < 4; ++j) {
        Oacc[j][i] *= alpha;
        Pw[(i + hi8) * 80 + j * 16 + l15] = f2bf(pj[j]);  // C-layout -> LDS
      }
    }
    asm volatile("s_wait_dscnt 0" ::: "memory");          // P visible wave-wide

    // O += P * V   (P re-read in A-fragment layout)
#pragma unroll
    for (int ks = 0; ks < 2; ++ks) {
      Frag pa;
      pa.u[0] = ld4(&Pw[l15 * 80 + ks * 32 + off]);
      pa.u[1] = ld4(&Pw[l15 * 80 + ks * 32 + off + 16]);
#pragma unroll
      for (int j2 = 0; j2 < 4; ++j2) {
        Frag vb;
        int drow = j2 * 16 + l15;
        int tof  = ks * 32 + off;
        vb.u[0] = ld4(&Vsm[drow * 80 + tof]);
        vb.u[1] = ld4(&Vsm[drow * 80 + tof + 16]);
        Oacc[j2] = WMMA_BF16(pa.v, vb.v, Oacc[j2]);
      }
    }
  }

  // Finalize: O /= l, store bf16 into [B,S,H*HD]
#pragma unroll
  for (int i = 0; i < 8; ++i) {
    float inv = 1.0f / l_i[i];
    int s = sbase + i + hi8;
#pragma unroll
    for (int j = 0; j < 4; ++j) {
      int col = h * HDm + j * 16 + l15;
      attnOut[((size_t)(b * Sm + s)) * Dm + col] = f2bf(Oacc[j][i] * inv);
    }
  }
}

// ---------------------------------------------------------------------------
// Host-side orchestration
// ---------------------------------------------------------------------------
extern "C" void kernel_launch(void* const* d_in, const int* in_sizes, int n_in,
                              void* d_out, int out_size, void* d_ws, size_t ws_size,
                              hipStream_t stream) {
  const float* x  = (const float*)d_in[0];   // [2,2048,1024]
  const float* Wq = (const float*)d_in[1];
  const float* bq = (const float*)d_in[2];
  const float* Wk = (const float*)d_in[3];
  const float* bk = (const float*)d_in[4];
  const float* Wv = (const float*)d_in[5];
  const float* bv = (const float*)d_in[6];
  const float* Wo = (const float*)d_in[7];
  const float* bo = (const float*)d_in[8];

  char* ws = (char*)d_ws;
  const size_t MB = 1u << 20;
  unsigned short* Xbf  = (unsigned short*)(ws + 0);        // 8 MB
  unsigned short* WqT  = (unsigned short*)(ws + 8  * MB);  // 2 MB each
  unsigned short* WkT  = (unsigned short*)(ws + 10 * MB);
  unsigned short* WvT  = (unsigned short*)(ws + 12 * MB);
  unsigned short* WoT  = (unsigned short*)(ws + 14 * MB);
  unsigned short* Qbf  = (unsigned short*)(ws + 16 * MB);  // 8 MB [B,H,S,HD]
  unsigned short* Kbf  = (unsigned short*)(ws + 24 * MB);  // 8 MB
  unsigned short* Vtbf = (unsigned short*)(ws + 32 * MB);  // 8 MB [B,H,HD,S]
  unsigned short* Attn = (unsigned short*)(ws + 0);        // reuse X region

  // Stage 1: conversions
  cvt_f32_bf16<<<(Mtot * Dm) / 256, 256, 0, stream>>>(x, Xbf);
  cvt_transpose_bf16<<<(Dm * Dm) / 256, 256, 0, stream>>>(Wq, WqT);
  cvt_transpose_bf16<<<(Dm * Dm) / 256, 256, 0, stream>>>(Wk, WkT);
  cvt_transpose_bf16<<<(Dm * Dm) / 256, 256, 0, stream>>>(Wv, WvT);
  cvt_transpose_bf16<<<(Dm * Dm) / 256, 256, 0, stream>>>(Wo, WoT);

  // Stage 2: QKV projections
  dim3 ggrid(Dm / 128, Mtot / 128);  // (8, 32)
  gemm_bf16_wmma<<<ggrid, 256, 0, stream>>>(Xbf, WqT, bq, (void*)Qbf,  0);
  gemm_bf16_wmma<<<ggrid, 256, 0, stream>>>(Xbf, WkT, bk, (void*)Kbf,  1);
  gemm_bf16_wmma<<<ggrid, 256, 0, stream>>>(Xbf, WvT, bv, (void*)Vtbf, 2);

  // Stage 3: attention (X region now dead -> reuse for attn output)
  dim3 agrid(Sm / 64, Bm * Hn);      // (32, 32)
  attn_wmma<<<agrid, 128, 0, stream>>>(Qbf, Kbf, Vtbf, Attn);

  // Stage 4: output projection -> f32 d_out
  gemm_bf16_wmma<<<ggrid, 256, 0, stream>>>(Attn, WoT, bo, d_out, 3);
}
